// privacyLoss2_79456894976223
// MI455X (gfx1250) — matrix operand order repaired and checked by reading, not compiled, not run on hardware
//
#include <hip/hip_runtime.h>
#include <hip/hip_bf16.h>

typedef __attribute__((ext_vector_type(16))) _Float16 v16h;
typedef __attribute__((ext_vector_type(8)))  float    v8f;
typedef int pl2_v4i __attribute__((vector_size(16)));   // matches builtin's expected pointee

#define KDIM   128
#define NWG    128           // gram workgroups (each stores one 128KB partial)
#define CHROWS 32            // batch rows per chunk (= WMMA K)
#define RSTRF  132           // padded f32 row stride in LDS (528B: 16B-aligned rows, low bank conflicts)

// ---- ws layout (floats) ----
// 0      G0 final [16384]
// 16384  G1 final [16384]
// 32768  s0[128]   32896 s1[128]
// 33024  n0   33025 n1   33026 q   33027 trace   33028 logdet1  33029 logdet2
// 33280  partials: WG w at 33280 + w*32768 : [Gtot_p 16384][G0_p 16384]
#define WS_S0   32768
#define WS_S1   32896
#define WS_SCAL 33024
#define WS_ZERO 33158
#define WS_PART 33280

// ---- async global->LDS staging (gfx1250 ASYNCcnt path), compile-safe fallback ----
#if defined(__HIP_DEVICE_COMPILE__) && defined(__gfx1250__) && \
    __has_builtin(__builtin_amdgcn_global_load_async_to_lds_b128) && \
    __has_builtin(__builtin_amdgcn_s_wait_asynccnt)
#define PL2_ASYNC 1
#else
#define PL2_ASYNC 0
#endif

#if PL2_ASYNC
#define PL2_AS1(p) ((__attribute__((address_space(1))) pl2_v4i*)(uintptr_t)(p))
#define PL2_AS3(p) ((__attribute__((address_space(3))) pl2_v4i*)(unsigned)(uintptr_t)(p))
#endif

__global__ void pl2_zero_kernel(float* ws, int n) {
    int i = blockIdx.x * 256 + threadIdx.x;
    if (i < n) ws[i] = 0.0f;
}

// stage one 32x128 f32 chunk (+32 labels) into LDS buffer `par`
__device__ __forceinline__ void pl2_stage(float* rawP, long long* lblP,
                                          const float* __restrict__ feat,
                                          const long long* __restrict__ label,
                                          int chunk, int tid) {
    const size_t rowBase = (size_t)chunk * CHROWS;
    const float* src = feat + rowBase * KDIM;
#if PL2_ASYNC
#pragma unroll
    for (int q = 0; q < 4; ++q) {
        int g = tid + q * 256;                              // float4 index 0..1023
        const float* gp = src + g * 4;
        float* lp = rawP + (g >> 5) * RSTRF + (g & 31) * 4; // 16B-aligned LDS dst
        __builtin_amdgcn_global_load_async_to_lds_b128(PL2_AS1(gp), PL2_AS3(lp), 0, 0);
    }
    if (tid < 16) {
        const long long* gp = label + rowBase + tid * 2;
        long long* lp = lblP + tid * 2;
        __builtin_amdgcn_global_load_async_to_lds_b128(PL2_AS1(gp), PL2_AS3(lp), 0, 0);
    }
#else
    const float4* s4 = (const float4*)src;
#pragma unroll
    for (int q = 0; q < 4; ++q) {
        int g = tid + q * 256;
        float4 v4 = s4[g];
        *(float4*)(rawP + (g >> 5) * RSTRF + (g & 31) * 4) = v4;
    }
    if (tid < CHROWS) lblP[tid] = label[rowBase + tid];
#endif
}

__device__ __forceinline__ void pl2_wait_stage(bool hasNext) {
#if PL2_ASYNC
    if (hasNext) __builtin_amdgcn_s_wait_asynccnt(4);   // prev batch drained, newest may remain
    else         __builtin_amdgcn_s_wait_asynccnt(0);
#else
    (void)hasNext;
#endif
}

__global__ __launch_bounds__(256) void pl2_gram_kernel(
        const float* __restrict__ feat, const long long* __restrict__ label,
        float* __restrict__ ws, int nChunks) {
    __shared__ float     raw[2][CHROWS * RSTRF];   // f32 chunk, double-buffered (33792 B)
    __shared__ long long lbl[2][CHROWS];           // labels, double-buffered (512 B)
    __shared__ unsigned  fragA[8 * 32 * 9];        // packed f16 A frags, stride 9 dwords (9216 B)
    __shared__ int       smask[CHROWS];            // 1 if label==0

    const int tid  = threadIdx.x;
    const int lane = tid & 31;
    const int wave = tid >> 5;                     // 0..7, owns output rows 16*wave..16*wave+15

    v8f accT[8], acc0[8];                          // Gtot and G0 accumulators (128 VGPRs)
    const v8f zf = {0.f,0.f,0.f,0.f,0.f,0.f,0.f,0.f};
#pragma unroll
    for (int j = 0; j < 8; ++j) { accT[j] = zf; acc0[j] = zf; }

    float s0a = 0.f, s1a = 0.f;                    // per-feature sums (tid<128, feature=tid)
    int cnt0 = 0, myChunks = 0;

    int par = 0;
    pl2_stage(&raw[0][0], &lbl[0][0], feat, label, blockIdx.x, tid);   // prologue

    for (int c = blockIdx.x; c < nChunks; c += gridDim.x, par ^= 1) {
        const int  cn      = c + gridDim.x;
        const bool hasNext = cn < nChunks;
        __syncthreads();                           // all waves done reading buf[par^1]
        if (hasNext) pl2_stage(&raw[par ^ 1][0], &lbl[par ^ 1][0], feat, label, cn, tid);
        pl2_wait_stage(hasNext);                   // own async batch for buf[par] complete
        __syncthreads();                           // cross-wave: buf[par] fully staged
        const float*     rc = &raw[par][0];
        const long long* lc = &lbl[par][0];
        // ---- pack A fragments (ISA 16-bit A 16x32 layout), f32 -> packed f16 via cvt_pkrtz ----
        {
            int v = tid >> 5, l = tid & 31;
            int fl = l & 15, half = l >> 4;
            int K0 = 2 * (v & 3) + 16 * (v >> 2) + 8 * half;
#pragma unroll
            for (int p = 0; p < 8; ++p) {
                float a = rc[K0 * RSTRF + p * 16 + fl];
                float b = rc[(K0 + 1) * RSTRF + p * 16 + fl];
                fragA[(p * 32 + l) * 9 + v] =
                    __builtin_bit_cast(unsigned, __builtin_amdgcn_cvt_pkrtz(a, b));
            }
            if (tid < CHROWS) smask[tid] = (lc[tid] == 0) ? 1 : 0;
        }
        // ---- per-class feature sums (f32) ----
        if (tid < KDIM) {
#pragma unroll 4
            for (int r = 0; r < CHROWS; ++r) {
                float x = rc[r * RSTRF + tid];
                if (lc[r] == 0) s0a += x; else s1a += x;
            }
        }
        if (tid == 0) { int cc = 0; for (int r = 0; r < CHROWS; ++r) cc += (lc[r] == 0); cnt0 += cc; }
        ++myChunks;
        __syncthreads();
        // ---- WMMA: wave w has A=frag(w); loop B over 8 col tiles; Gtot unmasked + G0 masked ----
        {
            union { unsigned u[8]; v16h h; } af;
#pragma unroll
            for (int v = 0; v < 8; ++v) af.u[v] = fragA[(wave * 32 + lane) * 9 + v];
            const int m = smask[lane];             // B-lane == batch row k
#pragma unroll
            for (int j = 0; j < 8; ++j) {
                union { unsigned u[8]; v16h h; } bt, b0;
                const float* br = rc + j * 16 + (lane & 15);
#pragma unroll
                for (int v = 0; v < 8; ++v) {
                    int k0 = 2 * (v & 3) + 16 * (v >> 2) + 8 * (lane >> 4);
                    float x = br[k0 * RSTRF], y = br[(k0 + 1) * RSTRF];
                    bt.u[v] = __builtin_bit_cast(unsigned, __builtin_amdgcn_cvt_pkrtz(x, y));
                    b0.u[v] = __builtin_bit_cast(unsigned, __builtin_amdgcn_cvt_pkrtz(smask[k0] ? x : 0.f, smask[k0 + 1] ? y : 0.f));
                }
                accT[j] = __builtin_amdgcn_wmma_f32_16x16x32_f16(false, af.h, false, bt.h,
                                                                 (short)0, accT[j], false, false);
                acc0[j] = __builtin_amdgcn_wmma_f32_16x16x32_f16(false, af.h, false, b0.h,
                                                                 (short)0, acc0[j], false, false);
            }
        }
    }
    // ---- store per-WG partials (no atomics on the big matrices) ----
    float* base = ws + WS_PART + (size_t)blockIdx.x * 32768;
    {
        int nn = lane & 15;
        int mtop = (lane >> 4) * 8;                // C/D layout: lanes16-31 hold M=8..15
#pragma unroll
        for (int j = 0; j < 8; ++j)
#pragma unroll
            for (int mi = 0; mi < 8; ++mi) {
                int row = wave * 16 + mtop + mi;
                int col = j * 16 + nn;
                base[row * KDIM + col]         = accT[j][mi];
                base[16384 + row * KDIM + col] = acc0[j][mi];
            }
    }
    if (tid < KDIM) { atomicAdd(ws + WS_S0 + tid, s0a); atomicAdd(ws + WS_S1 + tid, s1a); }
    if (tid == 0) {
        atomicAdd(ws + WS_SCAL + 0, (float)cnt0);
        atomicAdd(ws + WS_SCAL + 1, (float)(myChunks * CHROWS - cnt0));
    }
}

__global__ void pl2_reduce_kernel(float* ws) {
    int i = blockIdx.x * 256 + threadIdx.x;        // 0..16383
    float st = 0.f, s0 = 0.f;
    const float* p = ws + WS_PART + i;
    for (int w = 0; w < NWG; ++w) {
        st += p[(size_t)w * 32768];
        s0 += p[(size_t)w * 32768 + 16384];
    }
    ws[i]         = s0;        // G0
    ws[16384 + i] = st - s0;   // G1 = Gtot - G0
}

__device__ __forceinline__ void pl2_chol128(float (*A)[KDIM], int t) {
    for (int k = 0; k < KDIM; ++k) {
        if (t == k) A[k][k] = sqrtf(A[k][k]);
        __syncthreads();
        if (t > k) A[t][k] /= A[k][k];
        __syncthreads();
        if (t > k) {
            float aik = A[t][k];
            for (int j = k + 1; j <= t; ++j) A[t][j] -= aik * A[j][k];
        }
        __syncthreads();
    }
}

__global__ __launch_bounds__(128) void pl2_final_kernel(float* ws, float* out) {
    __shared__ float A[KDIM][KDIM];                // exactly 64 KB
    const int t = threadIdx.x;
    const float* G0 = ws;
    const float* G1 = ws + 16384;
    const float* s0 = ws + WS_S0;
    const float* s1 = ws + WS_S1;
    const float n0 = ws[WS_SCAL + 0], n1 = ws[WS_SCAL + 1];
    float* q_acc  = ws + WS_SCAL + 2;
    float* tr_acc = ws + WS_SCAL + 3;
    float* ld1    = ws + WS_SCAL + 4;
    float* ld2    = ws + WS_SCAL + 5;

    // ---- build Sf2 (lower triangle); both classes centered at mu1 = s0/n0 (faithful bug) ----
    {
        float mui = s0[t] / n0, s1i = s1[t];
        for (int j = 0; j <= t; ++j) {
            float muj = s0[j] / n0;
            float val = G1[t * KDIM + j] - mui * s1[j] - s1i * muj + n1 * mui * muj;
            A[t][j] = ((t == j) ? 1.0f : 0.0f) + val / n1;
        }
    }
    __syncthreads();
    pl2_chol128(A, t);                             // A lower = L2
    atomicAdd(ld2, 2.0f * logf(A[t][t]));
    __syncthreads();
    float drec = 1.0f / A[t][t];
    __syncthreads();
    A[t][t] = drec;                                // diag = 1/L[k][k] = T[k][k]
    __syncthreads();
    // ---- triangular inverse T = L2^-1, T^T stored in strictly-upper (race-free) ----
    {
        int j = t;
        for (int i = j + 1; i < KDIM; ++i) {
            float s = A[i][j] * A[j][j];           // L[i][j] * T[j][j]
            for (int k = j + 1; k < i; ++k) s += A[i][k] * A[j][k];   // L[i][k]*T[k][j]
            A[j][i] = -s * A[i][i];                // T[i][j]
        }
    }
    __syncthreads();
    // ---- q = d^T Sf2^-1 d = ||T d||^2 ; d recomputed on the fly ----
    {
        float dm = s0[t] / n0 - s1[t] / n1;
        float ym = A[t][t] * dm;
        for (int j = 0; j < t; ++j) {
            float dj = s0[j] / n0 - s1[j] / n1;
            ym += A[j][t] * dj;                    // T[t][j] = A[j][t]
        }
        atomicAdd(q_acc, ym * ym);
    }
    // ---- trace(Sf2^-1 Sf1) = sum_{i>=j} w * inv2[i][j] * Sf1[i][j] ----
    {
        float tl = 0.f;
        int cnt = 0;
        for (int i = 0; i < KDIM; ++i)
            for (int j = 0; j <= i; ++j, ++cnt) {
                if ((cnt & 127) != t) continue;
                float s = 0.f;
                for (int m = i; m < KDIM; ++m) {
                    float Tmi = (m == i) ? A[i][i] : A[i][m];
                    float Tmj = (m == j) ? A[j][j] : A[j][m];
                    s += Tmi * Tmj;
                }
                float mui = s0[i] / n0, muj = s0[j] / n0;
                float sf = ((i == j) ? 1.0f : 0.0f) +
                    (G0[i * KDIM + j] - mui * s0[j] - s0[i] * muj + n0 * mui * muj) / n0;
                tl += ((i == j) ? 1.0f : 2.0f) * s * sf;
            }
        atomicAdd(tr_acc, tl);
    }
    __syncthreads();
    // ---- rebuild Sf1, Cholesky for logdet1 ----
    {
        float mui = s0[t] / n0, s0i = s0[t];
        for (int j = 0; j <= t; ++j) {
            float muj = s0[j] / n0;
            float val = G0[t * KDIM + j] - mui * s0[j] - s0i * muj + n0 * mui * muj;
            A[t][j] = ((t == j) ? 1.0f : 0.0f) + val / n0;
        }
    }
    __syncthreads();
    pl2_chol128(A, t);
    atomicAdd(ld1, 2.0f * logf(A[t][t]));
    __threadfence();
    __syncthreads();
    if (t == 0) {
        float v2 = atomicAdd(ld2, 0.0f);
        float v1 = atomicAdd(ld1, 0.0f);
        float vq = atomicAdd(q_acc, 0.0f);
        float vt = atomicAdd(tr_acc, 0.0f);
        out[0] = 0.5f * ((v2 - v1) * 1.4426950408889634f - (float)KDIM + vq + vt);
    }
}

extern "C" void kernel_launch(void* const* d_in, const int* in_sizes, int n_in,
                              void* d_out, int out_size, void* d_ws, size_t ws_size,
                              hipStream_t stream) {
    const float*     feat  = (const float*)d_in[0];
    const long long* label = (const long long*)d_in[1];
    float* out = (float*)d_out;
    float* ws  = (float*)d_ws;
    int Btot    = in_sizes[0] / KDIM;          // 262144
    int nChunks = Btot / CHROWS;               // 8192

    pl2_zero_kernel<<<(WS_ZERO + 255) / 256, 256, 0, stream>>>(ws, WS_ZERO);
    pl2_gram_kernel<<<NWG, 256, 0, stream>>>(feat, label, ws, nChunks);
    pl2_reduce_kernel<<<16384 / 256, 256, 0, stream>>>(ws);
    pl2_final_kernel<<<1, 128, 0, stream>>>(ws, out);
}
